// GruDecoder_66228395704834
// MI455X (gfx1250) — compile-verified, matrix operand order checked by reading, not run on hardware
//
#include <hip/hip_runtime.h>
#include <hip/hip_bf16.h>

// GRU decoder scan: T=256 steps, B=256, HID=1024, OUT=768.
// Persistent kernel + grid barriers (serial scan => avoid 512 launches).
// bf16 WMMA (v_wmma_f32_16x16x32_bf16), fp32 accumulate; weights converted once
// to bf16, served from L2 every step. Register-blocked 32x32 per wave:
// 12 WMMAs per 16 b128 loads per k-iteration.

#define T_LEN   256
#define BATCH   256
#define HID     1024
#define OUT_DIM 768
#define NBLOCKS 32
#define NTHREADS 256

typedef __bf16 bf16_t;
typedef __attribute__((ext_vector_type(16))) __bf16 v16bf;
typedef __attribute__((ext_vector_type(8)))  __bf16 v8bf;
typedef __attribute__((ext_vector_type(8)))  float  v8f;

union V16u { v16bf v; v8bf h[2]; };

// A-fragment (16x32, row-major act) and B-fragment (32x16, weights stored [N][K])
// have identical per-lane load shape: two contiguous 16B loads from lane's row.
__device__ __forceinline__ v16bf ldfrag(const bf16_t* rowPtr, int k, int kHalf) {
  V16u u;
  u.h[0] = *(const v8bf*)(rowPtr + k + kHalf);
  u.h[1] = *(const v8bf*)(rowPtr + k + kHalf + 16);
  return u.v;
}

__device__ __forceinline__ v8f wmma_bf16(v16bf a, v16bf b, v8f c) {
  return __builtin_amdgcn_wmma_f32_16x16x32_bf16(false, a, false, b, (short)0, c,
                                                 false, false);
}

__device__ __forceinline__ float sigmoidf_(float x) {
  return 1.0f / (1.0f + __expf(-x));
}

// Grid-wide generation barrier (persistent-kernel phase sync).
__device__ __forceinline__ void grid_barrier(unsigned* cnt, unsigned* gen,
                                             unsigned target) {
  __syncthreads();
  if (threadIdx.x == 0) {
    __threadfence();
    if (atomicAdd(cnt, 1u) == (unsigned)gridDim.x - 1u) {
      *cnt = 0u;
      __threadfence();
      atomicExch(gen, target);
    } else {
      while (atomicAdd(gen, 0u) < target) { __builtin_amdgcn_s_sleep(2); }
    }
    __threadfence();
  }
  __syncthreads();
}

// fp32 -> bf16 conversion of weights + initial x0/h0, and barrier-state reset.
__global__ void gru_init_kernel(const float* __restrict__ src,
                                const float* __restrict__ hidden,
                                const float* __restrict__ W_ih,
                                const float* __restrict__ W_hh,
                                const float* __restrict__ W_fc,
                                bf16_t* __restrict__ xbuf,
                                bf16_t* __restrict__ h0buf,
                                bf16_t* __restrict__ wih,
                                bf16_t* __restrict__ whh,
                                bf16_t* __restrict__ wfc,
                                unsigned* __restrict__ bar) {
  size_t i = (size_t)blockIdx.x * blockDim.x + threadIdx.x;
  size_t stride = (size_t)gridDim.x * blockDim.x;
  if (i == 0) { bar[0] = 0u; bar[1] = 0u; }
  for (size_t j = i; j < (size_t)3 * HID * OUT_DIM; j += stride) wih[j] = (bf16_t)W_ih[j];
  for (size_t j = i; j < (size_t)3 * HID * HID;     j += stride) whh[j] = (bf16_t)W_hh[j];
  for (size_t j = i; j < (size_t)OUT_DIM * HID;     j += stride) wfc[j] = (bf16_t)W_fc[j];
  for (size_t j = i; j < (size_t)BATCH * OUT_DIM;   j += stride) xbuf[j]  = (bf16_t)src[j];    // src[0]
  for (size_t j = i; j < (size_t)BATCH * HID;       j += stride) h0buf[j] = (bf16_t)hidden[j]; // hidden[0]
}

__global__ void __launch_bounds__(NTHREADS)
gru_decoder_kernel(const float* __restrict__ b_ih,
                   const float* __restrict__ b_hh,
                   const float* __restrict__ b_fc,
                   const bf16_t* __restrict__ wih,   // [3H][OUT]
                   const bf16_t* __restrict__ whh,   // [3H][HID]
                   const bf16_t* __restrict__ wfc,   // [OUT][HID]
                   bf16_t* __restrict__ xbuf,        // [B][OUT]  (y fed back)
                   bf16_t* __restrict__ h0buf,       // [B][HID]  ping
                   bf16_t* __restrict__ h1buf,       // [B][HID]  pong
                   float* __restrict__ out,          // [T][B][OUT]
                   unsigned* __restrict__ bar) {
  unsigned* cnt = bar + 0;
  unsigned* gen = bar + 1;

  const int lane  = threadIdx.x & 31;
  const int wave  = threadIdx.x >> 5;   // 8 waves: 4 (M) x 2 (N)
  const int wm    = wave & 3;           // 32-row wave sub-tile
  const int wn    = wave >> 2;          // 32-col wave sub-tile
  const int laneN = lane & 15;
  const int kHalf = (lane >> 4) << 3;   // A/B fragment K half (0 or 8)
  const int mOff  = (lane >> 4) << 3;   // C/D: lanes 16-31 hold M=v+8

  // Phase-1 tiling: h_new [256 x 1024] -> block tile 128x64 -> 2 x 16 = 32 blocks.
  const int p1_tm = blockIdx.x & 1;
  const int p1_tn = blockIdx.x >> 1;
  const int p1_rBase = p1_tm * 128 + wm * 32;   // wave owns rows [rBase, rBase+32)
  const int p1_cBase = p1_tn * 64 + wn * 32;    // wave owns cols [cBase, cBase+32)

  for (int t = 0; t < T_LEN; ++t) {
    bf16_t* hPrev = (t & 1) ? h1buf : h0buf;
    bf16_t* hNext = (t & 1) ? h0buf : h1buf;

    // ---------------- Phase 1: h_new = GRU(x, hPrev) ----------------
    {
      // [i][j]: i = M sub-tile (16 rows), j = N sub-tile (16 cols)
      v8f accR[2][2]  = {};  // x@Wih_r^T + h@Whh_r^T (fused)
      v8f accZ[2][2]  = {};  // x@Wih_z^T + h@Whh_z^T (fused)
      v8f accIN[2][2] = {};  // x@Wih_n^T
      v8f accHN[2][2] = {};  // h@Whh_n^T

      const bf16_t* aP0 = xbuf + (size_t)(p1_rBase + laneN) * OUT_DIM;
      const bf16_t* aP1 = xbuf + (size_t)(p1_rBase + 16 + laneN) * OUT_DIM;
      {
        const bf16_t* bR0 = wih + (size_t)(p1_cBase + laneN) * OUT_DIM;
        const bf16_t* bR1 = wih + (size_t)(p1_cBase + 16 + laneN) * OUT_DIM;
        const bf16_t* bZ0 = wih + (size_t)(HID + p1_cBase + laneN) * OUT_DIM;
        const bf16_t* bZ1 = wih + (size_t)(HID + p1_cBase + 16 + laneN) * OUT_DIM;
        const bf16_t* bN0 = wih + (size_t)(2 * HID + p1_cBase + laneN) * OUT_DIM;
        const bf16_t* bN1 = wih + (size_t)(2 * HID + p1_cBase + 16 + laneN) * OUT_DIM;
        for (int k = 0; k < OUT_DIM; k += 32) {
          __builtin_prefetch((const void*)(aP0 + k + 64), 0, 3);  // WGP-scope
          v16bf a0 = ldfrag(aP0, k, kHalf);
          v16bf a1 = ldfrag(aP1, k, kHalf);
          v16bf b;
          b = ldfrag(bR0, k, kHalf);
          accR[0][0]  = wmma_bf16(a0, b, accR[0][0]);
          accR[1][0]  = wmma_bf16(a1, b, accR[1][0]);
          b = ldfrag(bR1, k, kHalf);
          accR[0][1]  = wmma_bf16(a0, b, accR[0][1]);
          accR[1][1]  = wmma_bf16(a1, b, accR[1][1]);
          b = ldfrag(bZ0, k, kHalf);
          accZ[0][0]  = wmma_bf16(a0, b, accZ[0][0]);
          accZ[1][0]  = wmma_bf16(a1, b, accZ[1][0]);
          b = ldfrag(bZ1, k, kHalf);
          accZ[0][1]  = wmma_bf16(a0, b, accZ[0][1]);
          accZ[1][1]  = wmma_bf16(a1, b, accZ[1][1]);
          b = ldfrag(bN0, k, kHalf);
          accIN[0][0] = wmma_bf16(a0, b, accIN[0][0]);
          accIN[1][0] = wmma_bf16(a1, b, accIN[1][0]);
          b = ldfrag(bN1, k, kHalf);
          accIN[0][1] = wmma_bf16(a0, b, accIN[0][1]);
          accIN[1][1] = wmma_bf16(a1, b, accIN[1][1]);
        }
      }
      const bf16_t* hP0 = hPrev + (size_t)(p1_rBase + laneN) * HID;
      const bf16_t* hP1 = hPrev + (size_t)(p1_rBase + 16 + laneN) * HID;
      {
        const bf16_t* cR0 = whh + (size_t)(p1_cBase + laneN) * HID;
        const bf16_t* cR1 = whh + (size_t)(p1_cBase + 16 + laneN) * HID;
        const bf16_t* cZ0 = whh + (size_t)(HID + p1_cBase + laneN) * HID;
        const bf16_t* cZ1 = whh + (size_t)(HID + p1_cBase + 16 + laneN) * HID;
        const bf16_t* cN0 = whh + (size_t)(2 * HID + p1_cBase + laneN) * HID;
        const bf16_t* cN1 = whh + (size_t)(2 * HID + p1_cBase + 16 + laneN) * HID;
        for (int k = 0; k < HID; k += 32) {
          __builtin_prefetch((const void*)(hP0 + k + 64), 0, 3);
          v16bf a0 = ldfrag(hP0, k, kHalf);
          v16bf a1 = ldfrag(hP1, k, kHalf);
          v16bf b;
          b = ldfrag(cR0, k, kHalf);
          accR[0][0]  = wmma_bf16(a0, b, accR[0][0]);
          accR[1][0]  = wmma_bf16(a1, b, accR[1][0]);
          b = ldfrag(cR1, k, kHalf);
          accR[0][1]  = wmma_bf16(a0, b, accR[0][1]);
          accR[1][1]  = wmma_bf16(a1, b, accR[1][1]);
          b = ldfrag(cZ0, k, kHalf);
          accZ[0][0]  = wmma_bf16(a0, b, accZ[0][0]);
          accZ[1][0]  = wmma_bf16(a1, b, accZ[1][0]);
          b = ldfrag(cZ1, k, kHalf);
          accZ[0][1]  = wmma_bf16(a0, b, accZ[0][1]);
          accZ[1][1]  = wmma_bf16(a1, b, accZ[1][1]);
          b = ldfrag(cN0, k, kHalf);
          accHN[0][0] = wmma_bf16(a0, b, accHN[0][0]);
          accHN[1][0] = wmma_bf16(a1, b, accHN[1][0]);
          b = ldfrag(cN1, k, kHalf);
          accHN[0][1] = wmma_bf16(a0, b, accHN[0][1]);
          accHN[1][1] = wmma_bf16(a1, b, accHN[1][1]);
        }
      }
      // Gate math + h_new writeback (C/D layout: N=laneN, M=v+mOff).
      #pragma unroll
      for (int i = 0; i < 2; ++i) {
        #pragma unroll
        for (int j = 0; j < 2; ++j) {
          const int col = p1_cBase + j * 16 + laneN;
          const float br  = b_ih[col] + b_hh[col];
          const float bz  = b_ih[HID + col] + b_hh[HID + col];
          const float bin = b_ih[2 * HID + col];
          const float bhn = b_hh[2 * HID + col];
          #pragma unroll
          for (int v = 0; v < 8; ++v) {
            const int row = p1_rBase + i * 16 + v + mOff;
            const float r  = sigmoidf_(accR[i][j][v] + br);
            const float z  = sigmoidf_(accZ[i][j][v] + bz);
            const float nn = tanhf(accIN[i][j][v] + bin + r * (accHN[i][j][v] + bhn));
            const float hp = (float)hPrev[(size_t)row * HID + col];
            const float hn = (1.0f - z) * nn + z * hp;
            hNext[(size_t)row * HID + col] = (bf16_t)hn;
          }
        }
      }
    }

    grid_barrier(cnt, gen, 2u * (unsigned)t + 1u);   // h_new complete

    // ---------------- Phase 2: y = h_new @ W_fc^T + b_fc ----------------
    // y [256 x 768] -> block tile 128x64 -> 2 x 12 = 24 tiles over 32 blocks.
    for (int tile = blockIdx.x; tile < 24; tile += NBLOCKS) {
      const int tm = tile & 1;
      const int tn = tile >> 1;
      const int rBase = tm * 128 + wm * 32;
      const int cBase = tn * 64 + wn * 32;
      v8f acc[2][2] = {};
      const bf16_t* aP0 = hNext + (size_t)(rBase + laneN) * HID;
      const bf16_t* aP1 = hNext + (size_t)(rBase + 16 + laneN) * HID;
      const bf16_t* b0 = wfc + (size_t)(cBase + laneN) * HID;
      const bf16_t* b1 = wfc + (size_t)(cBase + 16 + laneN) * HID;
      for (int k = 0; k < HID; k += 32) {
        v16bf a0 = ldfrag(aP0, k, kHalf);
        v16bf a1 = ldfrag(aP1, k, kHalf);
        v16bf b;
        b = ldfrag(b0, k, kHalf);
        acc[0][0] = wmma_bf16(a0, b, acc[0][0]);
        acc[1][0] = wmma_bf16(a1, b, acc[1][0]);
        b = ldfrag(b1, k, kHalf);
        acc[0][1] = wmma_bf16(a0, b, acc[0][1]);
        acc[1][1] = wmma_bf16(a1, b, acc[1][1]);
      }
      float* outT = out + (size_t)t * BATCH * OUT_DIM;
      #pragma unroll
      for (int i = 0; i < 2; ++i) {
        #pragma unroll
        for (int j = 0; j < 2; ++j) {
          const int col = cBase + j * 16 + laneN;
          const float bias = b_fc[col];
          #pragma unroll
          for (int v = 0; v < 8; ++v) {
            const int row = rBase + i * 16 + v + mOff;
            const float y = acc[i][j][v] + bias;
            outT[(size_t)row * OUT_DIM + col] = y;           // fp32 result
            xbuf[(size_t)row * OUT_DIM + col] = (bf16_t)y;   // feedback input
          }
        }
      }
    }

    grid_barrier(cnt, gen, 2u * (unsigned)t + 2u);   // y / x-feedback complete
  }
}

extern "C" void kernel_launch(void* const* d_in, const int* in_sizes, int n_in,
                              void* d_out, int out_size, void* d_ws, size_t ws_size,
                              hipStream_t stream) {
  // setup_inputs order: src, tgt, hidden, W_ih, W_hh, b_ih, b_hh, W_fc, b_fc
  const float* src    = (const float*)d_in[0];
  // d_in[1] = tgt (int64) -> only its length (256) matters; compile-time constant.
  const float* hidden = (const float*)d_in[2];
  const float* W_ih   = (const float*)d_in[3];
  const float* W_hh   = (const float*)d_in[4];
  const float* b_ih   = (const float*)d_in[5];
  const float* b_hh   = (const float*)d_in[6];
  const float* W_fc   = (const float*)d_in[7];
  const float* b_fc   = (const float*)d_in[8];
  float* out = (float*)d_out;

  // Workspace carve-out (~14 MB total), 256B-aligned slices.
  char* ws = (char*)d_ws;
  size_t off = 0;
  auto carve = [&](size_t bytes) -> char* {
    char* p = ws + off;
    off = (off + bytes + 255) & ~(size_t)255;
    return p;
  };
  bf16_t*   wih  = (bf16_t*)carve((size_t)3 * HID * OUT_DIM * sizeof(bf16_t));
  bf16_t*   whh  = (bf16_t*)carve((size_t)3 * HID * HID * sizeof(bf16_t));
  bf16_t*   wfc  = (bf16_t*)carve((size_t)OUT_DIM * HID * sizeof(bf16_t));
  bf16_t*   xbuf = (bf16_t*)carve((size_t)BATCH * OUT_DIM * sizeof(bf16_t));
  bf16_t*   h0   = (bf16_t*)carve((size_t)BATCH * HID * sizeof(bf16_t));
  bf16_t*   h1   = (bf16_t*)carve((size_t)BATCH * HID * sizeof(bf16_t));
  unsigned* bar  = (unsigned*)carve(2 * sizeof(unsigned));

  gru_init_kernel<<<256, 256, 0, stream>>>(src, hidden, W_ih, W_hh, W_fc,
                                           xbuf, h0, wih, whh, wfc, bar);
  gru_decoder_kernel<<<NBLOCKS, NTHREADS, 0, stream>>>(b_ih, b_hh, b_fc,
                                                       wih, whh, wfc,
                                                       xbuf, h0, h1, out, bar);
}
